// LSTM_51281909514775
// MI455X (gfx1250) — matrix operand order repaired and checked by reading, not compiled, not run on hardware
//
#include <hip/hip_runtime.h>
#include <hip/hip_bf16.h>
#include <math.h>

typedef __attribute__((ext_vector_type(16))) __bf16 v16bf;
typedef __attribute__((ext_vector_type(8)))  __bf16 v8bf;
typedef __attribute__((ext_vector_type(8)))  float  v8f;
typedef unsigned int u32x4 __attribute__((ext_vector_type(4)));
typedef int vi4 __attribute__((vector_size(16)));   // matches builtin param type

#define LAGS  128
#define BATCH 1024
#define XDIM  256
#define HID   512
#define YDIM  128
#define NKH   (HID / 32)    // 16 K-blocks for H@W
#define NKX   (XDIM / 32)   // 8  K-blocks for X@U
#define RPAD  40            // LDS row stride in halves (32 data + 8 pad)

#if defined(__HIP_DEVICE_COMPILE__) && __has_builtin(__builtin_amdgcn_global_load_async_to_lds_b128)
#define ASYNC_STAGE 1
#else
#define ASYNC_STAGE 0
#endif

// ---------------- workspace layout (bytes) ----------------
#define OFF_XB  ((size_t)0)                               // X bf16 [T*B, XDIM]
#define OFF_UT  (OFF_XB + (size_t)LAGS*BATCH*XDIM*2)      // 4x Ut bf16 [HID, XDIM]
#define OFF_WT  (OFF_UT + (size_t)4*HID*XDIM*2)           // 4x Wt bf16 [HID, HID]
#define OFF_VT  (OFF_WT + (size_t)4*HID*HID*2)            // Vt bf16 [YDIM, HID]
#define OFF_H0  (OFF_VT + (size_t)YDIM*HID*2)             // H ping bf16 [B, HID]
#define OFF_H1  (OFF_H0 + (size_t)BATCH*HID*2)            // H pong bf16 [B, HID]
#define OFF_C   (OFF_H1 + (size_t)BATCH*HID*2)            // C f32 [B, HID]
#define OFF_HY  (OFF_C  + (size_t)BATCH*HID*4)            // Hy f32 [B, YDIM]

// ---------------- helpers ----------------
static __device__ __forceinline__ unsigned short f2bf(float f) {
  unsigned int u = __builtin_bit_cast(unsigned int, f);
  unsigned int r = u + 0x7FFFu + ((u >> 16) & 1u);   // round-to-nearest-even
  return (unsigned short)(r >> 16);
}
static __device__ __forceinline__ float sigmoidf_(float x) {
  return 1.0f / (1.0f + __expf(-x));
}
// A operand: lane holds row m0+(lane&15); K chunks at +0..7 / +16..23 (lane<16),
// +8..15 / +24..31 (lane>=16). Two 16-byte loads.
static __device__ __forceinline__ v16bf load_a16(const unsigned short* p) {
  union { v16bf v; v8bf h[2]; } u;
  u.h[0] = *(const v8bf*)(const void*)(p);
  u.h[1] = *(const v8bf*)(const void*)(p + 16);
  return u.v;
}
// B operand from two 8-half chunks at +0 and +16 (two 16B chunks so only 16B alignment needed; LDS-safe)
static __device__ __forceinline__ v16bf load_b16c(const unsigned short* p) {
  union { v16bf v; v8bf h[2]; } u;
  u.h[0] = *(const v8bf*)(const void*)(p);
  u.h[1] = *(const v8bf*)(const void*)(p + 16);
  return u.v;
}

#if ASYNC_STAGE
// 16 B per lane, global -> LDS, tracked by ASYNCcnt (no VGPR data round-trip)
static __device__ __forceinline__ void stage_async(const unsigned short* g, unsigned short* l) {
  __builtin_amdgcn_global_load_async_to_lds_b128(
      (vi4*)(void*)const_cast<unsigned short*>(g),
      (vi4*)(void*)l, 0, 0);
}
static __device__ __forceinline__ void wait_async0() {
#if __has_builtin(__builtin_amdgcn_s_wait_asynccnt)
  __builtin_amdgcn_s_wait_asynccnt(0);
#else
  asm volatile("s_wait_asynccnt 0x0" ::: "memory");
#endif
}
#endif

// ---------------- prep kernels ----------------
__global__ void k_cvt_bf16(const float* __restrict__ in, unsigned short* __restrict__ out, int n) {
  int i = blockIdx.x * blockDim.x + threadIdx.x;
  int stride = gridDim.x * blockDim.x;
  for (; i < n; i += stride) out[i] = f2bf(in[i]);
}
// in: f32 [K,N] row-major  ->  out: bf16 [N,K] row-major
__global__ void k_transpose_bf16(const float* __restrict__ in, unsigned short* __restrict__ out,
                                 int K, int N) {
  int i = blockIdx.x * blockDim.x + threadIdx.x;
  int total = K * N;
  int stride = gridDim.x * blockDim.x;
  for (; i < total; i += stride) {
    int n = i / K, k = i - n * K;
    out[i] = f2bf(in[k * N + n]);
  }
}
__global__ void k_zero_u32(unsigned int* __restrict__ p, int n) {
  int i = blockIdx.x * blockDim.x + threadIdx.x;
  int stride = gridDim.x * blockDim.x;
  for (; i < n; i += stride) p[i] = 0u;
}

// ---------------- LSTM step ----------------
// grid 256 x block 256 (8 waves). Block = 8 M-tiles x 1 N-tile; the four gates'
// B tiles are staged once per block into double-buffered LDS and shared by all waves.
__global__ __launch_bounds__(256) void k_lstm_step(
    const unsigned short* __restrict__ Hprev, unsigned short* __restrict__ Hnext,
    float* __restrict__ C, const unsigned short* __restrict__ Xt,
    const unsigned short* __restrict__ WtF, const unsigned short* __restrict__ WtI,
    const unsigned short* __restrict__ WtC, const unsigned short* __restrict__ WtO,
    const unsigned short* __restrict__ UtF, const unsigned short* __restrict__ UtI,
    const unsigned short* __restrict__ UtC, const unsigned short* __restrict__ UtO,
    const float* __restrict__ bF, const float* __restrict__ bI,
    const float* __restrict__ bC, const float* __restrict__ bO)
{
  __shared__ unsigned short sm[2][4][16 * RPAD];   // 2 bufs x 4 gates x 16 rows, ~10 KB

  const int tid  = threadIdx.x;
  const int lane = tid & 31;
  const int wave = tid >> 5;
  const int nt   = blockIdx.x & 31;   // N tile (shared by block)
  const int mg   = blockIdx.x >> 5;   // M group 0..7
  const int m0   = (mg * 8 + wave) * 16;
  const int n0   = nt * 16;
  const int half = lane >> 4;
  const int nl   = lane & 15;

  // staging decomposition: thread moves 8 halves (16 B) of one gate's B tile
  const int sIdx = tid * 8;           // 0..2040
  const int sG   = sIdx >> 9;         // gate 0..3
  const int sRow = (sIdx >> 5) & 15;  // row-in-tile 0..15
  const int sK   = sIdx & 31;         // 0,8,16,24
  const unsigned short* BgW = (sG == 0) ? WtF : (sG == 1) ? WtI : (sG == 2) ? WtC : WtO;
  const unsigned short* BgU = (sG == 0) ? UtF : (sG == 1) ? UtI : (sG == 2) ? UtC : UtO;
  unsigned short* sDst[2] = { &sm[0][sG][sRow * RPAD + sK], &sm[1][sG][sRow * RPAD + sK] };

  v8f aF = {}, aI = {}, aC = {}, aO = {};
  int buf = 0;
  const int ldsOff = nl * RPAD + half * 8;

  // ---- phase 1: gates += H @ W  (K = 512) ----
  {
    const unsigned short* aBase = Hprev + (m0 + nl) * HID + half * 8;
    const unsigned short* sSrc  = BgW + (n0 + sRow) * HID + sK;
#if ASYNC_STAGE
    stage_async(sSrc, sDst[buf]);
    for (int kb = 0; kb < NKH; ++kb) {
      wait_async0();                  // current buffer landed (async loads complete in order)
      __syncthreads();                // ..and all waves' prior LDS reads are done
      if (kb + 1 < NKH) stage_async(sSrc + (kb + 1) * 32, sDst[buf ^ 1]);
#else
    u32x4 r = *(const u32x4*)(const void*)(sSrc);
    for (int kb = 0; kb < NKH; ++kb) {
      *(u32x4*)(void*)(sDst[buf]) = r;
      if (kb + 1 < NKH) r = *(const u32x4*)(const void*)(sSrc + (kb + 1) * 32);
      __syncthreads();
#endif
      v16bf a  = load_a16(aBase + kb * 32);
      v16bf b0 = load_b16c(&sm[buf][0][ldsOff]);
      v16bf b1 = load_b16c(&sm[buf][1][ldsOff]);
      v16bf b2 = load_b16c(&sm[buf][2][ldsOff]);
      v16bf b3 = load_b16c(&sm[buf][3][ldsOff]);
      aF = __builtin_amdgcn_wmma_f32_16x16x32_bf16(false, a, false, b0, (short)0, aF, false, false);
      aI = __builtin_amdgcn_wmma_f32_16x16x32_bf16(false, a, false, b1, (short)0, aI, false, false);
      aC = __builtin_amdgcn_wmma_f32_16x16x32_bf16(false, a, false, b2, (short)0, aC, false, false);
      aO = __builtin_amdgcn_wmma_f32_16x16x32_bf16(false, a, false, b3, (short)0, aO, false, false);
      buf ^= 1;
    }
  }
  // ---- phase 2: gates += X_t @ U  (K = 256) ----
  {
    const unsigned short* aBase = Xt + (m0 + nl) * XDIM + half * 8;
    const unsigned short* sSrc  = BgU + (n0 + sRow) * XDIM + sK;
#if ASYNC_STAGE
    stage_async(sSrc, sDst[buf]);
    for (int kb = 0; kb < NKX; ++kb) {
      wait_async0();
      __syncthreads();
      if (kb + 1 < NKX) stage_async(sSrc + (kb + 1) * 32, sDst[buf ^ 1]);
#else
    u32x4 r = *(const u32x4*)(const void*)(sSrc);
    for (int kb = 0; kb < NKX; ++kb) {
      *(u32x4*)(void*)(sDst[buf]) = r;
      if (kb + 1 < NKX) r = *(const u32x4*)(const void*)(sSrc + (kb + 1) * 32);
      __syncthreads();
#endif
      v16bf a  = load_a16(aBase + kb * 32);
      v16bf b0 = load_b16c(&sm[buf][0][ldsOff]);
      v16bf b1 = load_b16c(&sm[buf][1][ldsOff]);
      v16bf b2 = load_b16c(&sm[buf][2][ldsOff]);
      v16bf b3 = load_b16c(&sm[buf][3][ldsOff]);
      aF = __builtin_amdgcn_wmma_f32_16x16x32_bf16(false, a, false, b0, (short)0, aF, false, false);
      aI = __builtin_amdgcn_wmma_f32_16x16x32_bf16(false, a, false, b1, (short)0, aI, false, false);
      aC = __builtin_amdgcn_wmma_f32_16x16x32_bf16(false, a, false, b2, (short)0, aC, false, false);
      aO = __builtin_amdgcn_wmma_f32_16x16x32_bf16(false, a, false, b3, (short)0, aO, false, false);
      buf ^= 1;
    }
  }

  // epilogue: C/D layout -> row = m0 + r + 8*half, col = n0 + (lane&15)
  const int col = n0 + nl;
  const float bFv = bF[col], bIv = bI[col], bCv = bC[col], bOv = bO[col];
  #pragma unroll
  for (int r = 0; r < 8; ++r) {
    const int row = m0 + r + half * 8;
    const int idx = row * HID + col;
    float f  = sigmoidf_(aF[r] + bFv);
    float ii = sigmoidf_(aI[r] + bIv);
    float g  = tanhf(aC[r] + bCv);
    float o  = sigmoidf_(aO[r] + bOv);
    float cn = f * C[idx] + ii * g;
    C[idx] = cn;
    Hnext[idx] = f2bf(o * tanhf(cn));
  }
}

// ---------------- head: Hy = H @ V + bias_out ----------------
__global__ __launch_bounds__(256) void k_head(
    const unsigned short* __restrict__ H, const unsigned short* __restrict__ Vt,
    const float* __restrict__ biasOut, float* __restrict__ Hy)
{
  const int lane = threadIdx.x & 31;
  const int wave = threadIdx.x >> 5;
  const int tile = blockIdx.x * 8 + wave;   // 0..511
  const int mt = tile >> 3, nt = tile & 7;
  const int m0 = mt * 16, n0 = nt * 16;
  const int half = lane >> 4;
  const int nl = lane & 15;

  v8f acc = {};
  const unsigned short* aH = H + (m0 + nl) * HID + half * 8;
  const int brow = n0 + nl;
  for (int k0 = 0; k0 < HID; k0 += 32) {
    v16bf a = load_a16(aH + k0);
    v16bf b = load_b16c(Vt + brow * HID + k0 + half * 8);
    acc = __builtin_amdgcn_wmma_f32_16x16x32_bf16(false, a, false, b, (short)0, acc, false, false);
  }
  const int col = n0 + nl;
  const float bv = biasOut[col];
  #pragma unroll
  for (int r = 0; r < 8; ++r) {
    const int row = m0 + r + half * 8;
    Hy[row * YDIM + col] = acc[r] + bv;
  }
}

// ---------------- out = Hy @ fc2_w^T + fc2_b ----------------
__global__ void k_out(const float* __restrict__ Hy, const float* __restrict__ W2,
                      const float* __restrict__ b2, float* __restrict__ out)
{
  int i = blockIdx.x * blockDim.x + threadIdx.x;   // 0..3071
  if (i >= BATCH * 3) return;
  int b = i / 3, j = i - b * 3;
  float s = b2[j];
  #pragma unroll 4
  for (int y = 0; y < YDIM; ++y) s += Hy[b * YDIM + y] * W2[j * YDIM + y];
  out[i] = s;
}

extern "C" void kernel_launch(void* const* d_in, const int* in_sizes, int n_in,
                              void* d_out, int out_size, void* d_ws, size_t ws_size,
                              hipStream_t stream) {
  (void)in_sizes; (void)n_in; (void)out_size; (void)ws_size;
  const float* X       = (const float*)d_in[0];
  const float* Wm[4]   = {(const float*)d_in[1], (const float*)d_in[2],
                          (const float*)d_in[3], (const float*)d_in[4]};
  const float* Um[4]   = {(const float*)d_in[5], (const float*)d_in[6],
                          (const float*)d_in[7], (const float*)d_in[8]};
  const float* bm[4]   = {(const float*)d_in[9], (const float*)d_in[10],
                          (const float*)d_in[11], (const float*)d_in[12]};
  const float* V       = (const float*)d_in[13];
  const float* biasOut = (const float*)d_in[14];
  const float* fc2w    = (const float*)d_in[15];
  const float* fc2b    = (const float*)d_in[16];

  char* ws = (char*)d_ws;
  unsigned short* Xb = (unsigned short*)(ws + OFF_XB);
  unsigned short* Ut[4], *Wt[4];
  for (int g = 0; g < 4; ++g) {
    Ut[g] = (unsigned short*)(ws + OFF_UT + (size_t)g * HID * XDIM * 2);
    Wt[g] = (unsigned short*)(ws + OFF_WT + (size_t)g * HID * HID * 2);
  }
  unsigned short* Vt = (unsigned short*)(ws + OFF_VT);
  unsigned short* H0 = (unsigned short*)(ws + OFF_H0);
  unsigned short* H1 = (unsigned short*)(ws + OFF_H1);
  float* C  = (float*)(ws + OFF_C);
  float* Hy = (float*)(ws + OFF_HY);

  // prep: bf16 conversions / transposes, zero init of state
  k_cvt_bf16<<<4096, 256, 0, stream>>>(X, Xb, LAGS * BATCH * XDIM);
  for (int g = 0; g < 4; ++g)
    k_transpose_bf16<<<512, 256, 0, stream>>>(Um[g], Ut[g], XDIM, HID);
  for (int g = 0; g < 4; ++g)
    k_transpose_bf16<<<1024, 256, 0, stream>>>(Wm[g], Wt[g], HID, HID);
  k_transpose_bf16<<<256, 256, 0, stream>>>(V, Vt, HID, YDIM);
  k_zero_u32<<<1024, 256, 0, stream>>>((unsigned int*)H0, BATCH * HID / 2);
  k_zero_u32<<<2048, 256, 0, stream>>>((unsigned int*)C, BATCH * HID);

  // sequential recurrence: stream ordering carries the dependency
  for (int t = 0; t < LAGS; ++t) {
    const unsigned short* hp = (t & 1) ? H1 : H0;
    unsigned short*       hn = (t & 1) ? H0 : H1;
    k_lstm_step<<<256, 256, 0, stream>>>(hp, hn, C,
        Xb + (size_t)t * BATCH * XDIM,
        Wt[0], Wt[1], Wt[2], Wt[3],
        Ut[0], Ut[1], Ut[2], Ut[3],
        bm[0], bm[1], bm[2], bm[3]);
  }
  // after 128 steps (even), final H lives in H0
  k_head<<<64, 256, 0, stream>>>(H0, Vt, biasOut, Hy);
  k_out<<<12, 256, 0, stream>>>(Hy, fc2w, fc2b, (float*)d_out);
}
